// WaveletDilationBlock_32650341384310
// MI455X (gfx1250) — matrix operand order repaired and checked by reading, not compiled, not run on hardware
//
#include <hip/hip_runtime.h>
#include <stdint.h>

typedef __attribute__((ext_vector_type(16))) __bf16 v16bf;
typedef __attribute__((ext_vector_type(8)))  float  v8f;

#define SQRT_HALF 0.70710678118654752440f

#if defined(__has_builtin)
#if __has_builtin(__builtin_amdgcn_global_load_async_to_lds_b128) && \
    __has_builtin(__builtin_amdgcn_s_wait_asynccnt)
#define HAVE_ASYNC 1
#endif
#endif
#ifndef HAVE_ASYNC
#define HAVE_ASYNC 0
#endif

#if HAVE_ASYNC
typedef int v4i_vs __attribute__((vector_size(16)));
typedef __attribute__((address_space(1))) v4i_vs* as1_v4i;
typedef __attribute__((address_space(3))) v4i_vs* as3_v4i;
#endif

struct PartialSums { double s1, s2; };

__device__ inline unsigned int f32_to_bf16_bits(float f) {
  unsigned int u = __float_as_uint(f);
  unsigned int r = u + 0x7fffu + ((u >> 16) & 1u);
  return r >> 16;
}

// ---------------------------------------------------------------------------
// Kernel 1: per-row 5-level Haar DWT in LDS + per-block |v| / v^2 partial sums
// ---------------------------------------------------------------------------
__global__ __launch_bounds__(256) void haar_stats_kernel(
    const float* __restrict__ x, float* __restrict__ coef,
    PartialSums* __restrict__ part) {
  __shared__ float cur[4096];
  __shared__ float nxt[2048];
  __shared__ float outc[4096];
  __shared__ double r1[256];
  __shared__ double r2[256];
  const int tid = threadIdx.x;
  const int row = blockIdx.x;
  const float* xr = x + (size_t)row * 4096;
  for (int i = tid; i < 4096; i += 256) cur[i] = xr[i];
  __syncthreads();
  float* src = cur;
  float* dst = nxt;
  int n = 4096;
  for (int lvl = 0; lvl < 5; ++lvl) {
    int half = n >> 1;
    for (int i = tid; i < half; i += 256) {
      float2 eo = *(const float2*)&src[2 * i];  // one ds_load_b64
      outc[half + i] = (eo.x - eo.y) * SQRT_HALF;  // detail at offset == half
      dst[i] = (eo.x + eo.y) * SQRT_HALF;
    }
    __syncthreads();
    float* t = src; src = dst; dst = t;
    n = half;
  }
  if (tid < 128) outc[tid] = src[tid];  // cA5 -> [0,128)
  __syncthreads();

  double s1 = 0.0, s2 = 0.0;
  float* cr = coef + (size_t)row * 4096;
  for (int i = tid; i < 4096; i += 256) {
    float v = outc[i];
    cr[i] = v;
    double dv = (double)v;
    s1 += fabs(dv);
    s2 += dv * dv;
  }
  r1[tid] = s1; r2[tid] = s2;
  __syncthreads();
  for (int off = 128; off > 0; off >>= 1) {
    if (tid < off) { r1[tid] += r1[tid + off]; r2[tid] += r2[tid + off]; }
    __syncthreads();
  }
  if (tid == 0) { part[row].s1 = r1[0]; part[row].s2 = r2[0]; }
}

// ---------------------------------------------------------------------------
// Kernel 2: split block_mat (f32, [K=4096][N=512]) into transposed bf16
// hi/lo planes Bt[n][k] so B fragments are contiguous-K reads.
// ---------------------------------------------------------------------------
__global__ __launch_bounds__(256) void convert_b_kernel(
    const float* __restrict__ Bm, unsigned short* __restrict__ Bh,
    unsigned short* __restrict__ Bl) {
  int idx = blockIdx.x * 256 + threadIdx.x;  // 0 .. 2097151
  int k = idx >> 9;
  int nn = idx & 511;
  float f = Bm[idx];
  unsigned int hs = f32_to_bf16_bits(f);
  float hf = __uint_as_float(hs << 16);
  unsigned int ls = f32_to_bf16_bits(f - hf);
  size_t o = (size_t)nn * 4096 + k;
  Bh[o] = (unsigned short)hs;
  Bl[o] = (unsigned short)ls;
}

// ---------------------------------------------------------------------------
// Kernel 3: deterministic reduction of partials -> threshold (ddof=1)
// ---------------------------------------------------------------------------
__global__ __launch_bounds__(256) void finalize_kernel(
    const PartialSums* __restrict__ part, float* __restrict__ thr) {
  __shared__ double r1[256];
  __shared__ double r2[256];
  const int tid = threadIdx.x;
  double s1 = 0.0, s2 = 0.0;
  for (int i = tid; i < 8192; i += 256) { s1 += part[i].s1; s2 += part[i].s2; }
  r1[tid] = s1; r2[tid] = s2;
  __syncthreads();
  for (int off = 128; off > 0; off >>= 1) {
    if (tid < off) { r1[tid] += r1[tid + off]; r2[tid] += r2[tid + off]; }
    __syncthreads();
  }
  if (tid == 0) {
    const double N = 33554432.0;  // 8192*4096
    double mean = r1[0] / N;
    double var = (r2[0] - r1[0] * r1[0] / N) / (N - 1.0);
    if (var < 0.0) var = 0.0;
    *thr = (float)(mean + sqrt(var));
  }
}

// ---------------------------------------------------------------------------
// Kernel 4: in-place sparsify + split: f32 coeff -> packed {lo16=bf16_hi,
// hi16=bf16_lo} dword. Threshold applied on exact f32 (mask matches ref).
// ---------------------------------------------------------------------------
__global__ __launch_bounds__(256) void sparsify_kernel(
    float* __restrict__ A, const float* __restrict__ thrp) {
  const float thr = *thrp;
  size_t i = ((size_t)blockIdx.x * 256 + threadIdx.x) * 4;
  float4 v = *(const float4*)(A + i);
  float f[4] = {v.x, v.y, v.z, v.w};
  unsigned int p[4];
#pragma unroll
  for (int j = 0; j < 4; ++j) {
    float val = (fabsf(f[j]) > thr) ? f[j] : 0.0f;
    unsigned int hs = f32_to_bf16_bits(val);
    float hf = __uint_as_float(hs << 16);
    unsigned int ls = f32_to_bf16_bits(val - hf);
    p[j] = (ls << 16) | hs;
  }
  uint4 o; o.x = p[0]; o.y = p[1]; o.z = p[2]; o.w = p[3];
  *(uint4*)(A + i) = o;
}

// ---------------------------------------------------------------------------
// Kernel 5: bf16x3 WMMA GEMM with fused ReLU.
// Block tile 128(M)x128(N), 8 waves, wave tile 64x32 (4x2 frags), K-step 32.
// Double-buffered LDS pipeline. A: global->VGPR, hi/lo v_perm split at LDS
// store. B: CDNA5 async global->LDS copy (ASYNCcnt) when available.
// ---------------------------------------------------------------------------
#define PADH 40  // halves per LDS row (32 + 8 pad, keeps 16B alignment)
#define LDSP (128 * PADH)

struct Tile {
  uint4 a[4];   // 16 consecutive packed dwords of one A row segment
  uint4 bh[2];  // used only in the non-async fallback
  uint4 bl[2];
};

__device__ __forceinline__ void load_a(Tile& t,
    const unsigned int* __restrict__ Apk, int m0, int kb, int tid) {
  // A: thread owns row r = tid>>1, K columns kc..kc+15 (kc = 0 or 16).
  int r = tid >> 1;
  int kc = (tid & 1) * 16;
  const unsigned int* gp = Apk + (size_t)(m0 + r) * 4096 + kb + kc;
#pragma unroll
  for (int j = 0; j < 4; ++j) t.a[j] = *(const uint4*)(gp + 4 * j);
}

__device__ __forceinline__ void stage_b(Tile& t,
    const unsigned short* __restrict__ Bh, const unsigned short* __restrict__ Bl,
    unsigned short* sBh, unsigned short* sBl, int n0, int kb, int tid) {
#pragma unroll
  for (int j = 0; j < 2; ++j) {
    int ch = tid + 256 * j;
    int r = ch >> 2;
    int c4 = ch & 3;
    size_t go = (size_t)(n0 + r) * 4096 + kb + c4 * 8;
    int so = r * PADH + c4 * 8;
#if HAVE_ASYNC
    __builtin_amdgcn_global_load_async_to_lds_b128(
        (as1_v4i)(Bh + go), (as3_v4i)(sBh + so), 0, 0);
    __builtin_amdgcn_global_load_async_to_lds_b128(
        (as1_v4i)(Bl + go), (as3_v4i)(sBl + so), 0, 0);
#else
    t.bh[j] = *(const uint4*)(Bh + go);
    t.bl[j] = *(const uint4*)(Bl + go);
    (void)so; (void)sBh; (void)sBl;
#endif
  }
}

__device__ __forceinline__ void store_a(const Tile& t,
    unsigned short* sAh, unsigned short* sAl, int tid) {
  // Unpack 16 packed dwords -> 8 hi dwords + 8 lo dwords (16 v_perm), store
  // as contiguous bf16 runs into the hi/lo A planes.
  int r = tid >> 1;
  int kc = (tid & 1) * 16;
  unsigned int H[8], L[8];
#pragma unroll
  for (int j = 0; j < 4; ++j) {
    H[2 * j]     = __builtin_amdgcn_perm(t.a[j].y, t.a[j].x, 0x05040100u);
    H[2 * j + 1] = __builtin_amdgcn_perm(t.a[j].w, t.a[j].z, 0x05040100u);
    L[2 * j]     = __builtin_amdgcn_perm(t.a[j].y, t.a[j].x, 0x07060302u);
    L[2 * j + 1] = __builtin_amdgcn_perm(t.a[j].w, t.a[j].z, 0x07060302u);
  }
  int off = r * PADH + kc;
  uint4 h0; h0.x = H[0]; h0.y = H[1]; h0.z = H[2]; h0.w = H[3];
  uint4 h1; h1.x = H[4]; h1.y = H[5]; h1.z = H[6]; h1.w = H[7];
  uint4 l0; l0.x = L[0]; l0.y = L[1]; l0.z = L[2]; l0.w = L[3];
  uint4 l1; l1.x = L[4]; l1.y = L[5]; l1.z = L[6]; l1.w = L[7];
  *(uint4*)&sAh[off] = h0;
  *(uint4*)&sAh[off + 8] = h1;
  *(uint4*)&sAl[off] = l0;
  *(uint4*)&sAl[off + 8] = l1;
}

__device__ __forceinline__ void store_b(const Tile& t,
    unsigned short* sBh, unsigned short* sBl, int tid) {
#if !HAVE_ASYNC
#pragma unroll
  for (int j = 0; j < 2; ++j) {
    int ch = tid + 256 * j;
    int r = ch >> 2;
    int c4 = ch & 3;
    *(uint4*)&sBh[r * PADH + c4 * 8] = t.bh[j];
    *(uint4*)&sBl[r * PADH + c4 * 8] = t.bl[j];
  }
#else
  (void)t; (void)sBh; (void)sBl; (void)tid;
#endif
}

__device__ __forceinline__ void wait_async() {
#if HAVE_ASYNC
  __builtin_amdgcn_s_wait_asynccnt(0);
#endif
}

__device__ __forceinline__ void compute_tile(
    const unsigned short* sAh, const unsigned short* sAl,
    const unsigned short* sBh, const unsigned short* sBl, v8f c[4][2],
    int waveM, int waveN, int l16, int hi16) {
  union B32 { uint4 q[2]; v16bf v; };

  // B fragments: lane n=l16, lane<16 K-runs {0..7,16..23}, lane>=16 {8..15,24..31}.
  v16bf bhf[2], blf[2];
#pragma unroll
  for (int fn = 0; fn < 2; ++fn) {
    int n = waveN * 32 + fn * 16 + l16;
    int off = n * PADH + hi16 * 8;
    B32 uh, ul;
    uh.q[0] = *(const uint4*)&sBh[off];
    uh.q[1] = *(const uint4*)&sBh[off + 16];
    ul.q[0] = *(const uint4*)&sBl[off];
    ul.q[1] = *(const uint4*)&sBl[off + 16];
    bhf[fn] = uh.v;
    blf[fn] = ul.v;
  }

  // A fragments: row m=l16 within frag; lane<16 K-runs {0..7,16..23},
  // lane>=16 K-runs {8..15,24..31}: two ds_load_b128 straight into operands.
  v16bf ahf[4], alf[4];
#pragma unroll
  for (int fm = 0; fm < 4; ++fm) {
    int m = waveM * 64 + fm * 16 + l16;
    int off = m * PADH + hi16 * 8;
    B32 uh, ul;
    uh.q[0] = *(const uint4*)&sAh[off];
    uh.q[1] = *(const uint4*)&sAh[off + 16];
    ul.q[0] = *(const uint4*)&sAl[off];
    ul.q[1] = *(const uint4*)&sAl[off + 16];
    ahf[fm] = uh.v;
    alf[fm] = ul.v;
  }

  // bf16x3 accumulation: Ahi@Bhi + Ahi@Blo + Alo@Bhi
#pragma unroll
  for (int fm = 0; fm < 4; ++fm) {
#pragma unroll
    for (int fn = 0; fn < 2; ++fn) {
      c[fm][fn] = __builtin_amdgcn_wmma_f32_16x16x32_bf16(
          false, ahf[fm], false, bhf[fn], (short)0, c[fm][fn], false, false);
      c[fm][fn] = __builtin_amdgcn_wmma_f32_16x16x32_bf16(
          false, ahf[fm], false, blf[fn], (short)0, c[fm][fn], false, false);
      c[fm][fn] = __builtin_amdgcn_wmma_f32_16x16x32_bf16(
          false, alf[fm], false, bhf[fn], (short)0, c[fm][fn], false, false);
    }
  }
}

__global__ __launch_bounds__(256) void gemm_kernel(
    const unsigned int* __restrict__ Apk, const unsigned short* __restrict__ Bh,
    const unsigned short* __restrict__ Bl, float* __restrict__ out) {
  __shared__ unsigned short sAh[2][LDSP];
  __shared__ unsigned short sAl[2][LDSP];
  __shared__ unsigned short sBh[2][LDSP];
  __shared__ unsigned short sBl[2][LDSP];

  const int tid = threadIdx.x;
  const int lane = tid & 31;
  const int wave = tid >> 5;
  const int waveM = wave & 1;
  const int waveN = wave >> 1;
  const int l16 = lane & 15;
  const int hi16 = lane >> 4;  // 0 or 1
  const int m0 = blockIdx.y * 128;
  const int n0 = blockIdx.x * 128;

  v8f c[4][2];
#pragma unroll
  for (int fm = 0; fm < 4; ++fm)
#pragma unroll
    for (int fn = 0; fn < 2; ++fn)
      c[fm][fn] = (v8f){0.f, 0.f, 0.f, 0.f, 0.f, 0.f, 0.f, 0.f};

  Tile t;
  // Prologue: tile 0 -> buf0
  stage_b(t, Bh, Bl, sBh[0], sBl[0], n0, 0, tid);
  load_a(t, Apk, m0, 0, tid);
  store_a(t, sAh[0], sAl[0], tid);
  store_b(t, sBh[0], sBl[0], tid);
  wait_async();
  __syncthreads();

  // Steady state: one barrier per tile; staging overlaps WMMA compute.
  // Async targets buffer X only after the barrier retiring all reads of X.
  for (int kb = 0; kb < 4096 - 64; kb += 64) {
    stage_b(t, Bh, Bl, sBh[1], sBl[1], n0, kb + 32, tid);
    load_a(t, Apk, m0, kb + 32, tid);
    compute_tile(sAh[0], sAl[0], sBh[0], sBl[0], c, waveM, waveN, l16, hi16);
    store_a(t, sAh[1], sAl[1], tid);
    store_b(t, sBh[1], sBl[1], tid);
    wait_async();
    __syncthreads();

    stage_b(t, Bh, Bl, sBh[0], sBl[0], n0, kb + 64, tid);
    load_a(t, Apk, m0, kb + 64, tid);
    compute_tile(sAh[1], sAl[1], sBh[1], sBl[1], c, waveM, waveN, l16, hi16);
    store_a(t, sAh[0], sAl[0], tid);
    store_b(t, sBh[0], sBl[0], tid);
    wait_async();
    __syncthreads();
  }
  // Epilogue pair: tiles 4032 (in buf0) and 4064.
  stage_b(t, Bh, Bl, sBh[1], sBl[1], n0, 4064, tid);
  load_a(t, Apk, m0, 4064, tid);
  compute_tile(sAh[0], sAl[0], sBh[0], sBl[0], c, waveM, waveN, l16, hi16);
  store_a(t, sAh[1], sAl[1], tid);
  store_b(t, sBh[1], sBl[1], tid);
  wait_async();
  __syncthreads();
  compute_tile(sAh[1], sAl[1], sBh[1], sBl[1], c, waveM, waveN, l16, hi16);

  // Epilogue: fused ReLU. C layout: VGPR r, lane l -> M=r+8*(l>=16), N=l%16.
#pragma unroll
  for (int fm = 0; fm < 4; ++fm) {
#pragma unroll
    for (int fn = 0; fn < 2; ++fn) {
      int n = n0 + waveN * 32 + fn * 16 + l16;
#pragma unroll
      for (int r = 0; r < 8; ++r) {
        int m = m0 + waveM * 64 + fm * 16 + r + 8 * hi16;
        out[(size_t)m * 512 + n] = fmaxf(c[fm][fn][r], 0.0f);
      }
    }
  }
}

// ---------------------------------------------------------------------------
extern "C" void kernel_launch(void* const* d_in, const int* in_sizes, int n_in,
                              void* d_out, int out_size, void* d_ws,
                              size_t ws_size, hipStream_t stream) {
  (void)in_sizes; (void)n_in; (void)out_size; (void)ws_size;
  const float* x = (const float*)d_in[0];
  const float* Bm = (const float*)d_in[1];
  float* out = (float*)d_out;
  char* ws = (char*)d_ws;

  const size_t OFF_A = 0;                        // 8192*4096*4 = 128 MiB
  const size_t OFF_BH = 134217728;               // 512*4096*2  = 4 MiB
  const size_t OFF_BL = OFF_BH + 4194304;        // 4 MiB
  const size_t OFF_P = OFF_BL + 4194304;         // 8192 * 16 B
  const size_t OFF_T = OFF_P + 8192 * sizeof(PartialSums);

  float* Acoef = (float*)(ws + OFF_A);
  unsigned short* Bh = (unsigned short*)(ws + OFF_BH);
  unsigned short* Bl = (unsigned short*)(ws + OFF_BL);
  PartialSums* part = (PartialSums*)(ws + OFF_P);
  float* thr = (float*)(ws + OFF_T);

  haar_stats_kernel<<<8192, 256, 0, stream>>>(x, Acoef, part);
  convert_b_kernel<<<8192, 256, 0, stream>>>(Bm, Bh, Bl);
  finalize_kernel<<<1, 256, 0, stream>>>(part, thr);
  sparsify_kernel<<<32768, 256, 0, stream>>>(Acoef, thr);
  dim3 grid(4, 64);  // x = N tiles (adjacent blocks share A rows -> L2 reuse)
  gemm_kernel<<<grid, 256, 0, stream>>>((const unsigned int*)Acoef, Bh, Bl, out);
}
